// SelfAttention_14190571946437
// MI455X (gfx1250) — matrix operand order repaired and checked by reading, not compiled, not run on hardware
//
#include <hip/hip_runtime.h>
#include <hip/hip_fp16.h>

typedef __attribute__((ext_vector_type(16))) _Float16 v16h;
typedef __attribute__((ext_vector_type(8)))  float    v8f;
typedef __attribute__((ext_vector_type(4)))  unsigned int v4u;
typedef __attribute__((ext_vector_type(8)))  int      v8i;
typedef __attribute__((ext_vector_type(4)))  int      v4i;

union Frag { v16h h; unsigned int u[8]; };

static __device__ __forceinline__ v8f wmma_f16(const Frag& a, const Frag& b, v8f c) {
    return __builtin_amdgcn_wmma_f32_16x16x32_f16(false, a.h, false, b.h,
                                                  (short)0, c, false, false);
}

// ---- TDM: 2D tile (f16) global -> LDS with row padding -------------------
// D# per CDNA5 ISA 8.3/8.4. tensor dims == tile dims (no OOB), data_size=2B.
// pad_interval code: 0=2,1=4,2=8,3=16,4=32... DWORDs; pad_amount code: n -> n+1 DWORDs.
static __device__ __forceinline__ void tdm_load_2d(unsigned lds_off, const void* gptr,
                                                   int tile_w, int tile_h,
                                                   int stride_elems,
                                                   int pad_i_code, int pad_a_code) {
    unsigned long long ga = (unsigned long long)gptr;
    v4u g0;
    g0[0] = 1u;                                        // count=1 (valid), user mode
    g0[1] = lds_off;                                   // lds_addr (bytes)
    g0[2] = (unsigned)ga;                              // global_addr[31:0]
    g0[3] = (unsigned)((ga >> 32) & 0x01FFFFFFu) | 0x80000000u;  // addr[56:32] | type=2
    v8i g1;
    g1[0] = (1 << 16) | (1 << 20) | (pad_i_code << 22) | (pad_a_code << 25);
    g1[1] = tile_w << 16;                              // tensor_dim0[15:0]
    g1[2] = tile_h << 16;                              // tensor_dim1[15:0]
    g1[3] = tile_w << 16;                              // tile_dim0
    g1[4] = tile_h & 0xFFFF;                           // tile_dim1 (tile_dim2=0 -> 2D)
    g1[5] = stride_elems;                              // tensor_dim0_stride[31:0]
    g1[6] = 0;
    g1[7] = 0;
    v4i z4 = {0, 0, 0, 0};
    v8i z8 = {0, 0, 0, 0, 0, 0, 0, 0};
    __builtin_amdgcn_tensor_load_to_lds(g0, g1, z4, z4, z8, 0);
}
static __device__ __forceinline__ unsigned lds_off(const void* p) {
    return (unsigned)(unsigned long long)p;            // generic->LDS offset = addr[31:0]
}

// ---- DPP16 butterfly reductions over 16-lane row groups (pure VALU) ----
static __device__ __forceinline__ float dpp_rowmax16(float x) {
    union { float f; int i; } u, t;
    u.f = x;
    t.i = __builtin_amdgcn_update_dpp(u.i, u.i, 0xB1,  0xF, 0xF, true); u.f = fmaxf(u.f, t.f);
    t.i = __builtin_amdgcn_update_dpp(u.i, u.i, 0x4E,  0xF, 0xF, true); u.f = fmaxf(u.f, t.f);
    t.i = __builtin_amdgcn_update_dpp(u.i, u.i, 0x141, 0xF, 0xF, true); u.f = fmaxf(u.f, t.f);
    t.i = __builtin_amdgcn_update_dpp(u.i, u.i, 0x140, 0xF, 0xF, true); u.f = fmaxf(u.f, t.f);
    return u.f;
}
static __device__ __forceinline__ float dpp_rowsum16(float x) {
    union { float f; int i; } u, t;
    u.f = x;
    t.i = __builtin_amdgcn_update_dpp(u.i, u.i, 0xB1,  0xF, 0xF, true); u.f += t.f;
    t.i = __builtin_amdgcn_update_dpp(u.i, u.i, 0x4E,  0xF, 0xF, true); u.f += t.f;
    t.i = __builtin_amdgcn_update_dpp(u.i, u.i, 0x141, 0xF, 0xF, true); u.f += t.f;
    t.i = __builtin_amdgcn_update_dpp(u.i, u.i, 0x140, 0xF, 0xF, true); u.f += t.f;
    return u.f;
}

// ---------------------------------------------------------------- cvt f32->f16
__global__ void cvt_f32_f16(const float* __restrict__ in, _Float16* __restrict__ out, int n) {
    int i = blockIdx.x * blockDim.x + threadIdx.x;
    if (i < n) out[i] = (_Float16)in[i];
}

// ---------------------------------------------------------------- GEMM (f16 in, f32 acc)
// BM=128, BN=64, BK=32, 256 threads (8 waves), wave grid 4x2, 2x2 WMMA tiles/wave.
// TDM double-buffered tiles: wave 0 DMAs tile k+1 while all waves compute tile k.
template <bool OUT_F16>
__global__ __launch_bounds__(256) void gemm_f16(const _Float16* __restrict__ A,
                                                const _Float16* __restrict__ B,
                                                const float* __restrict__ bias,
                                                void* __restrict__ Cout,
                                                int M, int N, int K) {
    __shared__ __align__(16) _Float16 As[2][128][40];   // 32 + 8 pad (pad: 16 dw -> +4 dw)
    __shared__ __align__(16) _Float16 Bs[2][32][72];    // 64 + 8 pad (pad: 32 dw -> +4 dw)

    const int tid   = threadIdx.x;
    const int lane  = tid & 31;
    const int wave  = tid >> 5;
    const int wm    = wave >> 1;
    const int wn    = wave & 1;
    const int l16   = lane & 15;
    const int lhalf = lane >> 4;
    const int m0    = blockIdx.y * 128;
    const int n0    = blockIdx.x * 64;

    v8f acc[2][2] = {};

    const int nk = K >> 5;
    if (wave == 0) {
        tdm_load_2d(lds_off(&As[0][0][0]), A + (size_t)m0 * K,      32, 128, K, 3, 3);
        tdm_load_2d(lds_off(&Bs[0][0][0]), B + n0,                  64,  32, N, 4, 3);
        __builtin_amdgcn_s_wait_tensorcnt(0);
    }
    __syncthreads();

    for (int kt = 0; kt < nk; ++kt) {
        const int buf = kt & 1;
        if (kt + 1 < nk && wave == 0) {
            const int k0 = (kt + 1) * 32;
            tdm_load_2d(lds_off(&As[buf ^ 1][0][0]), A + (size_t)m0 * K + k0, 32, 128, K, 3, 3);
            tdm_load_2d(lds_off(&Bs[buf ^ 1][0][0]), B + (size_t)k0 * N + n0, 64,  32, N, 4, 3);
        }

        Frag afr[2];
        #pragma unroll
        for (int ti = 0; ti < 2; ++ti) {
            int row = wm * 32 + ti * 16 + l16;
            #pragma unroll
            for (int v = 0; v < 8; ++v) {
                int kb = 2 * v + (v >= 4 ? 8 : 0) + 8 * lhalf;   // A-frag K pattern
                afr[ti].u[v] = *(const unsigned int*)&As[buf][row][kb];
            }
        }
        Frag bfr[2];
        #pragma unroll
        for (int nt = 0; nt < 2; ++nt) {
            int n = wn * 32 + nt * 16 + l16;
            #pragma unroll
            for (int v = 0; v < 8; ++v) {
                int kb = 2 * v + (v >= 4 ? 8 : 0) + 8 * lhalf;   // B-frag K pattern
                unsigned int lo = *(const unsigned short*)&Bs[buf][kb][n];
                unsigned int hi = *(const unsigned short*)&Bs[buf][kb + 1][n];
                bfr[nt].u[v] = lo | (hi << 16);
            }
        }
        #pragma unroll
        for (int ti = 0; ti < 2; ++ti)
            #pragma unroll
            for (int nt = 0; nt < 2; ++nt)
                acc[ti][nt] = wmma_f16(afr[ti], bfr[nt], acc[ti][nt]);

        __syncthreads();                       // all reads of buf complete
        if (kt + 1 < nk) {
            if (wave == 0) __builtin_amdgcn_s_wait_tensorcnt(0);
            __syncthreads();                   // buf^1 published
        }
    }

    #pragma unroll
    for (int ti = 0; ti < 2; ++ti) {
        #pragma unroll
        for (int nt = 0; nt < 2; ++nt) {
            int col = n0 + wn * 32 + nt * 16 + l16;
            float bv = bias ? bias[col] : 0.0f;
            #pragma unroll
            for (int r = 0; r < 8; ++r) {
                int row = m0 + wm * 32 + ti * 16 + r + 8 * lhalf;
                float v = acc[ti][nt][r] + bv;
                if (OUT_F16)
                    ((_Float16*)Cout)[(size_t)row * N + col] = (_Float16)v;
                else
                    ((float*)Cout)[(size_t)row * N + col] = v;
            }
        }
    }
}

// ---------------------------------------------------------------- flash attention
// qkv: [B=2, T=2048, 3072] f16 (q|k|v, 16 heads x 64). y: [B,T,1024] f16.
// 128 query rows/block, 8 waves x 16 rows. Key tiles of 32, TDM double-buffered.
__global__ __launch_bounds__(256) void attn_kernel(const _Float16* __restrict__ qkv,
                                                   _Float16* __restrict__ y) {
    constexpr int T = 2048, C3 = 3072;
    __shared__ __align__(16) _Float16 Qs[128][72];
    __shared__ __align__(16) _Float16 Ks[2][32][72];
    __shared__ __align__(16) _Float16 Vs[2][32][72];
    __shared__ __align__(16) _Float16 Ps[8][16][40];   // per-wave 16x32 P staging

    const int q0 = blockIdx.x * 128;
    const int h  = blockIdx.y;
    const int b  = blockIdx.z;
    const int tid = threadIdx.x;
    const int lane = tid & 31, wave = tid >> 5;
    const int l16 = lane & 15, lhalf = lane >> 4;

    const size_t base = (size_t)b * T * C3;
    const _Float16* Qg = qkv + base + h * 64;
    const _Float16* Kg = qkv + base + 1024 + h * 64;
    const _Float16* Vg = qkv + base + 2048 + h * 64;

    const int ntiles = q0 / 32 + 4;   // causal: keys only up to q0+127

    if (wave == 0) {
        tdm_load_2d(lds_off(&Qs[0][0]),       Qg + (size_t)q0 * C3, 64, 128, C3, 4, 3);
        tdm_load_2d(lds_off(&Ks[0][0][0]),    Kg,                   64,  32, C3, 4, 3);
        tdm_load_2d(lds_off(&Vs[0][0][0]),    Vg,                   64,  32, C3, 4, 3);
        __builtin_amdgcn_s_wait_tensorcnt(0);
    }
    __syncthreads();

    // Q A-fragments (rows wave*16..+15, two K-chunks of 32 over d=64)
    Frag qa[2];
    #pragma unroll
    for (int kc = 0; kc < 2; ++kc) {
        int row = wave * 16 + l16;
        #pragma unroll
        for (int v = 0; v < 8; ++v) {
            int kb = 2 * v + (v >= 4 ? 8 : 0) + 8 * lhalf + 32 * kc;
            qa[kc].u[v] = *(const unsigned int*)&Qs[row][kb];
        }
    }

    v8f o[4] = {};
    float mi[8], li[8];
    #pragma unroll
    for (int r = 0; r < 8; ++r) { mi[r] = -__builtin_inff(); li[r] = 0.0f; }

    for (int t = 0; t < ntiles; ++t) {
        const int j0  = t * 32;
        const int buf = t & 1;
        if (t + 1 < ntiles && wave == 0) {
            const size_t jn = (size_t)(t + 1) * 32;
            tdm_load_2d(lds_off(&Ks[buf ^ 1][0][0]), Kg + jn * C3, 64, 32, C3, 4, 3);
            tdm_load_2d(lds_off(&Vs[buf ^ 1][0][0]), Vg + jn * C3, 64, 32, C3, 4, 3);
        }

        // S = Q * K^T : load all four K-frags first, then 4 WMMAs
        Frag kf[2][2];
        #pragma unroll
        for (int nt = 0; nt < 2; ++nt) {
            int j = nt * 16 + l16;
            #pragma unroll
            for (int kc = 0; kc < 2; ++kc)
                #pragma unroll
                for (int v = 0; v < 8; ++v) {
                    int kb = 2 * v + (v >= 4 ? 8 : 0) + 8 * lhalf + 32 * kc;
                    kf[nt][kc].u[v] = *(const unsigned int*)&Ks[buf][j][kb];
                }
        }
        v8f s[2];
        #pragma unroll
        for (int nt = 0; nt < 2; ++nt) {
            s[nt] = (v8f)0.0f;
            #pragma unroll
            for (int kc = 0; kc < 2; ++kc)
                s[nt] = wmma_f16(qa[kc], kf[nt][kc], s[nt]);
        }

        // online softmax (causal; prefix-block mask is redundant with causal)
        float p0a[8], p1a[8];
        #pragma unroll
        for (int r = 0; r < 8; ++r) {
            int qi = q0 + wave * 16 + r + 8 * lhalf;
            float s0 = s[0][r] * 0.125f;
            float s1 = s[1][r] * 0.125f;
            if (j0 + l16 > qi)      s0 = -__builtin_inff();
            if (j0 + 16 + l16 > qi) s1 = -__builtin_inff();
            float mt    = dpp_rowmax16(fmaxf(s0, s1));
            float mnew  = fmaxf(mi[r], mt);
            float alpha = __expf(mi[r] - mnew);
            mi[r] = mnew;
            float p0 = __expf(s0 - mnew);
            float p1 = __expf(s1 - mnew);
            float rs = dpp_rowsum16(p0 + p1);
            li[r] = li[r] * alpha + rs;
            p0a[r] = p0; p1a[r] = p1;
            #pragma unroll
            for (int dt = 0; dt < 4; ++dt) o[dt][r] *= alpha;
        }

        // C-layout -> A-layout via per-wave LDS bounce
        #pragma unroll
        for (int r = 0; r < 8; ++r) {
            Ps[wave][r + 8 * lhalf][l16]      = (_Float16)p0a[r];
            Ps[wave][r + 8 * lhalf][16 + l16] = (_Float16)p1a[r];
        }
        asm volatile("s_wait_dscnt 0" ::: "memory");
        __builtin_amdgcn_wave_barrier();

        Frag pf;
        #pragma unroll
        for (int v = 0; v < 8; ++v) {
            int kb = 2 * v + (v >= 4 ? 8 : 0) + 8 * lhalf;
            pf.u[v] = *(const unsigned int*)&Ps[wave][l16][kb];
        }
        // O += P * V : load all four V-frags first, then 4 WMMAs
        Frag vf[4];
        #pragma unroll
        for (int dt = 0; dt < 4; ++dt) {
            int n = dt * 16 + l16;
            #pragma unroll
            for (int v = 0; v < 8; ++v) {
                int kb = 2 * v + (v >= 4 ? 8 : 0) + 8 * lhalf;
                unsigned int lo = *(const unsigned short*)&Vs[buf][kb][n];
                unsigned int hi = *(const unsigned short*)&Vs[buf][kb + 1][n];
                vf[dt].u[v] = lo | (hi << 16);
            }
        }
        #pragma unroll
        for (int dt = 0; dt < 4; ++dt)
            o[dt] = wmma_f16(pf, vf[dt], o[dt]);

        __syncthreads();                       // all reads of buf complete
        if (t + 1 < ntiles) {
            if (wave == 0) __builtin_amdgcn_s_wait_tensorcnt(0);
            __syncthreads();                   // buf^1 published
        }
    }

    // write y[b, t, h*64 + d] as f16 for projection GEMM
    #pragma unroll
    for (int dt = 0; dt < 4; ++dt) {
        #pragma unroll
        for (int r = 0; r < 8; ++r) {
            int qi = q0 + wave * 16 + r + 8 * lhalf;
            float val = o[dt][r] / li[r];
            y[((size_t)b * T + qi) * 1024 + h * 64 + dt * 16 + l16] = (_Float16)val;
        }
    }
}

// ---------------------------------------------------------------- launch
extern "C" void kernel_launch(void* const* d_in, const int* in_sizes, int n_in,
                              void* d_out, int out_size, void* d_ws, size_t ws_size,
                              hipStream_t stream) {
    const float* x      = (const float*)d_in[0];   // [2,2048,1024]
    const float* w_attn = (const float*)d_in[1];   // [1024,3072]
    const float* b_attn = (const float*)d_in[2];   // [3072]
    const float* w_proj = (const float*)d_in[3];   // [1024,1024]
    const float* b_proj = (const float*)d_in[4];   // [1024]
    float* out = (float*)d_out;                    // [2,2048,1024] f32

    char* ws = (char*)d_ws;
    _Float16* xh   = (_Float16*)(ws);                          //  8 MiB
    _Float16* wah  = (_Float16*)(ws + ((size_t)8  << 20));     //  6 MiB
    _Float16* wph  = (_Float16*)(ws + ((size_t)14 << 20));     //  2 MiB
    _Float16* qkvh = (_Float16*)(ws + ((size_t)16 << 20));     // 24 MiB
    _Float16* yh   = (_Float16*)(ws + ((size_t)40 << 20));     //  8 MiB

    const int nx  = 4096 * 1024;
    const int nwa = 1024 * 3072;
    const int nwp = 1024 * 1024;
    cvt_f32_f16<<<(nx  + 255) / 256, 256, 0, stream>>>(x,      xh,  nx);
    cvt_f32_f16<<<(nwa + 255) / 256, 256, 0, stream>>>(w_attn, wah, nwa);
    cvt_f32_f16<<<(nwp + 255) / 256, 256, 0, stream>>>(w_proj, wph, nwp);

    // QKV: [4096,1024] x [1024,3072] -> f16
    gemm_f16<true><<<dim3(3072 / 64, 4096 / 128), 256, 0, stream>>>(
        xh, wah, b_attn, qkvh, 4096, 3072, 1024);

    // attention: grid (q-tiles, heads, batch)
    attn_kernel<<<dim3(16, 16, 2), 256, 0, stream>>>(qkvh, yh);

    // projection: [4096,1024] x [1024,1024] -> f32 out
    gemm_f16<false><<<dim3(1024 / 64, 4096 / 128), 256, 0, stream>>>(
        yh, wph, b_proj, out, 4096, 1024, 1024);
}
